// EquivariantScalar_50337016709941
// MI455X (gfx1250) — matrix operand order repaired and checked by reading, not compiled, not run on hardware
//
#include <hip/hip_runtime.h>
#include <hip/hip_bf16.h>
#include <math.h>

// ---------------- problem constants ----------------
#define NF   128
#define NA   200000
#define BMOL 256
#define TA   16          // atoms per workgroup tile
#define NTHREADS 256     // 8 waves (wave32)

typedef __bf16 bf16;
typedef __attribute__((ext_vector_type(8)))  __bf16 bf16x8;
typedef __attribute__((ext_vector_type(16))) __bf16 bf16x16;
typedef __attribute__((ext_vector_type(8)))  float  f32x8;

// bf16 weight offsets inside d_ws (element units)
#define OFF_U0   0
#define OFF_V0   16384
#define OFF_W10  32768     // a0_w1: (128 out, 256 in)
#define OFF_W20  65536     // a0_w2: (256 out, 128 in)
#define OFF_U1   98304
#define OFF_V1   114688
#define OFF_W11  131072
#define OFF_W21  163840
#define W_TOTAL  196608    // bf16 elements
#define SOUT_BYTE_OFF ((size_t)W_TOTAL * 2)   // f32 s_out[NA] follows weights

// ---------------- fragment helpers (CDNA5 WMMA bf16 16x16x32 layouts) ----------------
// A (16x32, MxK): lane m (0..15): K 0..7 then 16..23 ; lane m+16: K 8..15 then 24..31
__device__ __forceinline__ bf16x16 a_frag(const bf16* tile, int stride, int kb) {
  const int lane = threadIdx.x & 31;
  const int m = lane & 15;
  const int o = (lane >> 4) << 3;                // 0 or 8
  const bf16* p = tile + m * stride + kb + o;
  bf16x8 lo = *(const bf16x8*)(p);               // K kb+o .. kb+o+7
  bf16x8 hi = *(const bf16x8*)(p + 16);          // K kb+16+o .. kb+23+o
  return __builtin_shufflevector(lo, hi, 0,1,2,3,4,5,6,7,8,9,10,11,12,13,14,15);
}

// B (32x16, KxN): lane n (0..15): K kb..kb+7 then kb+16..kb+23 of column N=n; lane n+16: K kb+8..kb+15 then kb+24..kb+31.
// W stored row-major (out=g, in=k) -> two 16B loads per lane.
__device__ __forceinline__ bf16x16 b_frag(const bf16* W, int Kdim, int gb, int kb) {
  const int lane = threadIdx.x & 31;
  const int n = lane & 15;
  const int k0 = kb + ((lane >> 4) << 3);        // +0 or +8
  const bf16* p = W + (gb + n) * Kdim + k0;
  bf16x8 lo = *(const bf16x8*)(p);
  bf16x8 hi = *(const bf16x8*)(p + 16);
  return __builtin_shufflevector(lo, hi, 0,1,2,3,4,5,6,7,8,9,10,11,12,13,14,15);
}

__device__ __forceinline__ f32x8 wmma_bf16(bf16x16 a, bf16x16 b, f32x8 c) {
  return __builtin_amdgcn_wmma_f32_16x16x32_bf16(false, a, false, b, (short)0, c, false, false);
}

// ---------------- weight pack: f32 -> bf16 ----------------
__global__ void pack_w(const float* __restrict__ src, bf16* __restrict__ dst, int n) {
  int i = blockIdx.x * blockDim.x + threadIdx.x;
  if (i < n) dst[i] = (bf16)src[i];
}

// ---------------- fused per-atom pipeline ----------------
__global__ __launch_bounds__(NTHREADS)
void eqscalar_main(const float* __restrict__ s_in, const float* __restrict__ v_in,
                   const bf16*  __restrict__ wts,
                   const float* __restrict__ b1_0, const float* __restrict__ b2_0,
                   const float* __restrict__ b1_1, const float* __restrict__ b2_1,
                   const float* __restrict__ out_w, const float* __restrict__ out_b,
                   float* __restrict__ s_out)
{
  // LDS: 8K + 12K + 12K + 4K + 4K = 40 KB
  __shared__ __align__(16) bf16 h  [TA * 2 * NF];   // [atom][0:128]=s, [128:256]=norm
  __shared__ __align__(16) bf16 vA [3 * TA * NF];   // [dir][atom][feat]
  __shared__ __align__(16) bf16 v1b[3 * TA * NF];   // gated-vector pre-product
  __shared__ __align__(16) bf16 hid[TA * NF];       // MLP hidden (post-SiLU)
  __shared__ __align__(16) bf16 ssg[TA * NF];       // vector gate

  const int tid  = threadIdx.x;
  const int lane = tid & 31;
  const int wid  = tid >> 5;
  const size_t abase = (size_t)blockIdx.x * TA;

  // ---- load s -> h[:,0:128], v -> vA[d][a][g] (coalesced, f32 -> bf16) ----
  for (int i = tid; i < TA * NF; i += NTHREADS) {
    int a = i >> 7, g = i & 127;
    h[a * 2 * NF + g] = (bf16)s_in[(abase + a) * NF + g];
  }
  for (int i = tid; i < 3 * TA * NF; i += NTHREADS) {
    int g = i & 127; int t = i >> 7;       // t = a*3 + d in global layout
    int a = t / 3, d = t - 3 * a;
    vA[(d * TA + a) * NF + g] = (bf16)v_in[(abase * 3) * NF + i];
  }
  __syncthreads();

  for (int l = 0; l < 2; ++l) {
    const bf16* WU = wts + (l ? OFF_U1  : OFF_U0);
    const bf16* WV = wts + (l ? OFF_V1  : OFF_V0);
    const bf16* W1 = wts + (l ? OFF_W11 : OFF_W10);
    const bf16* W2 = wts + (l ? OFF_W21 : OFF_W20);
    const float* B1 = l ? b1_1 : b1_0;
    const float* B2 = l ? b2_1 : b2_0;

    const int gb  = wid * 16;
    const int n   = lane & 15;
    const int hi8 = (lane >> 4) << 3;      // C/D layout: VGPR r -> row M = r + hi8

    // ---- stage 1+2 fused: v1 = v@uw^T, v2 = v@vw^T, norm(v2) -> h[:,128:256]
    //      Shared A fragments; B fragments hoisted (invariant across 3 dirs).
    {
      bf16x16 bv[4], bu[4];
      #pragma unroll
      for (int k = 0; k < 4; ++k) {
        bv[k] = b_frag(WV, NF, gb, k * 32);
        bu[k] = b_frag(WU, NF, gb, k * 32);
      }
      float acc[8] = {0.f,0.f,0.f,0.f,0.f,0.f,0.f,0.f};
      #pragma unroll
      for (int d = 0; d < 3; ++d) {
        const bf16* vt = vA + d * TA * NF;
        f32x8 c2 = {};     // v2 accumulator
        f32x8 c1 = {};     // v1 accumulator
        #pragma unroll
        for (int k = 0; k < 4; ++k) {
          bf16x16 af = a_frag(vt, NF, k * 32);
          c2 = wmma_bf16(af, bv[k], c2);
          c1 = wmma_bf16(af, bu[k], c1);
        }
        #pragma unroll
        for (int r = 0; r < 8; ++r) { float x = c2[r] + 1e-8f; acc[r] += x * x; }
        bf16* dst = v1b + d * TA * NF;
        #pragma unroll
        for (int r = 0; r < 8; ++r) dst[(r + hi8) * NF + gb + n] = (bf16)c1[r];
      }
      #pragma unroll
      for (int r = 0; r < 8; ++r)
        h[(r + hi8) * 2 * NF + NF + gb + n] = (bf16)sqrtf(acc[r]);
    }
    __syncthreads();

    // ---- stage 3: MLP1: silu(h @ W1^T + b1) -> hid (K = 256) ----
    {
      f32x8 c = {};
      for (int kb = 0; kb < 2 * NF; kb += 32)
        c = wmma_bf16(a_frag(h, 2 * NF, kb), b_frag(W1, 2 * NF, gb, kb), c);
      const float bias = B1[gb + n];
      #pragma unroll
      for (int r = 0; r < 8; ++r) {
        float x = c[r] + bias;
        x = x / (1.f + expf(-x));          // SiLU
        hid[(r + hi8) * NF + gb + n] = (bf16)x;
      }
    }
    __syncthreads();

    // ---- stage 4: MLP2: hid @ W2^T + b2 -> [sg | ss]  (K = 128, N = 256) ----
    for (int t = wid; t < 16; t += 8) {
      const int go = t * 16;
      f32x8 c = {};
      for (int kb = 0; kb < NF; kb += 32)
        c = wmma_bf16(a_frag(hid, NF, kb), b_frag(W2, NF, go, kb), c);
      const float bias = B2[go + n];
      if (t < 8) {                          // sg -> new s (low half of h)
        #pragma unroll
        for (int r = 0; r < 8; ++r) h[(r + hi8) * 2 * NF + go + n] = (bf16)(c[r] + bias);
      } else {                              // ss -> gate
        #pragma unroll
        for (int r = 0; r < 8; ++r) ssg[(r + hi8) * NF + (go - NF) + n] = (bf16)(c[r] + bias);
      }
    }
    __syncthreads();

    // ---- stage 5: v <- v1 * ss ----
    for (int i = tid; i < 3 * TA * NF; i += NTHREADS) {
      int g = i & 127; int a = (i >> 7) & 15;
      vA[i] = (bf16)((float)v1b[i] * (float)ssg[a * NF + g]);
    }
    __syncthreads();
  }

  // ---- final: s_out[a] = s . out_w + out_b ----
  if (tid < TA) {
    float acc = out_b[0];
    for (int g = 0; g < NF; ++g) acc += (float)h[tid * 2 * NF + g] * out_w[g];
    s_out[abase + tid] = acc;
  }
}

// ---------------- masked pool: y[b] = sum_a batch[b,a] * s_out[a] ----------------
__global__ __launch_bounds__(NTHREADS)
void pool_k(const float* __restrict__ batch, const float* __restrict__ sOut,
            float* __restrict__ y)
{
  __shared__ float red[NTHREADS];
  const int b = blockIdx.x;
  const float* row = batch + (size_t)b * NA;
  float acc = 0.f;
  for (int a = threadIdx.x; a < NA; a += NTHREADS) acc += row[a] * sOut[a];
  red[threadIdx.x] = acc;
  __syncthreads();
  for (int s = NTHREADS / 2; s > 0; s >>= 1) {
    if (threadIdx.x < s) red[threadIdx.x] += red[threadIdx.x + s];
    __syncthreads();
  }
  if (threadIdx.x == 0) y[b] = red[0];
}

// ---------------- entry ----------------
extern "C" void kernel_launch(void* const* d_in, const int* in_sizes, int n_in,
                              void* d_out, int out_size, void* d_ws, size_t ws_size,
                              hipStream_t stream) {
  const float* s     = (const float*)d_in[0];
  const float* v     = (const float*)d_in[1];
  // d_in[2] = pos (unused)
  const float* batch = (const float*)d_in[3];
  const float* u0    = (const float*)d_in[4];
  const float* v0    = (const float*)d_in[5];
  const float* w1_0  = (const float*)d_in[6];
  const float* b1_0  = (const float*)d_in[7];
  const float* w2_0  = (const float*)d_in[8];
  const float* b2_0  = (const float*)d_in[9];
  const float* u1    = (const float*)d_in[10];
  const float* v1w   = (const float*)d_in[11];
  const float* w1_1  = (const float*)d_in[12];
  const float* b1_1  = (const float*)d_in[13];
  const float* w2_1  = (const float*)d_in[14];
  const float* b2_1  = (const float*)d_in[15];
  const float* out_w = (const float*)d_in[16];
  const float* out_b = (const float*)d_in[17];

  bf16*  wts   = (bf16*)d_ws;
  float* s_out = (float*)((char*)d_ws + SOUT_BYTE_OFF);

  // pack weights to bf16 (every call; workspace is not persistent)
  pack_w<<<64,  NTHREADS, 0, stream>>>(u0,   wts + OFF_U0,  16384);
  pack_w<<<64,  NTHREADS, 0, stream>>>(v0,   wts + OFF_V0,  16384);
  pack_w<<<128, NTHREADS, 0, stream>>>(w1_0, wts + OFF_W10, 32768);
  pack_w<<<128, NTHREADS, 0, stream>>>(w2_0, wts + OFF_W20, 32768);
  pack_w<<<64,  NTHREADS, 0, stream>>>(u1,   wts + OFF_U1,  16384);
  pack_w<<<64,  NTHREADS, 0, stream>>>(v1w,  wts + OFF_V1,  16384);
  pack_w<<<128, NTHREADS, 0, stream>>>(w1_1, wts + OFF_W11, 32768);
  pack_w<<<128, NTHREADS, 0, stream>>>(w2_1, wts + OFF_W21, 32768);

  eqscalar_main<<<NA / TA, NTHREADS, 0, stream>>>(
      s, v, wts, b1_0, b2_0, b1_1, b2_1, out_w, out_b, s_out);

  pool_k<<<BMOL, NTHREADS, 0, stream>>>(batch, s_out, (float*)d_out);
}
